// IntraTemporalAttention_4183298146636
// MI455X (gfx1250) — matrix operand order repaired and checked by reading, not compiled, not run on hardware
//
#include <hip/hip_runtime.h>
#include <hip/hip_bf16.h>

#define TDIM 512
#define BATCH 32
#define HDIM 512
#define TEDIM 2048

typedef __bf16 v16bf __attribute__((ext_vector_type(16)));
typedef float  v8f   __attribute__((ext_vector_type(8)));

union Frag {
    v16bf v;
    uint4 q[2];
    unsigned int w[8];
    unsigned short u[16];
};

__device__ __forceinline__ unsigned int bfround(float f) {
    unsigned int x = __float_as_uint(f);
    return x + 0x7FFFu + ((x >> 16) & 1u);   // round-to-nearest-even bias
}
__device__ __forceinline__ unsigned int pack2(float lo, float hi) {
    return (bfround(lo) >> 16) | (bfround(hi) & 0xFFFF0000u);
}
__device__ __forceinline__ unsigned short bf1(float f) {
    return (unsigned short)(bfround(f) >> 16);
}

// ---- Pre-pass 1: enc (B,H,TE) f32 -> bf16, same layout ----
__global__ __launch_bounds__(256)
void conv_enc_kernel(const float* __restrict__ enc, unsigned short* __restrict__ out)
{
    const size_t i = (size_t)blockIdx.x * blockDim.x + threadIdx.x;  // 8 elements each
    const float4 f0 = ((const float4*)enc)[2 * i];
    const float4 f1 = ((const float4*)enc)[2 * i + 1];
    uint4 o;
    o.x = pack2(f0.x, f0.y);
    o.y = pack2(f0.z, f0.w);
    o.z = pack2(f1.x, f1.y);
    o.w = pack2(f1.z, f1.w);
    ((uint4*)out)[i] = o;
}

// ---- Pre-pass 2: enc (B,H,TE) f32 -> encT (B,TE,H) bf16 (tiled transpose) ----
__global__ __launch_bounds__(256)
void conv_encT_kernel(const float* __restrict__ enc, unsigned short* __restrict__ outT)
{
    __shared__ unsigned short tile[64][72];   // 64e x 64h, padded
    const int tid = threadIdx.x;
    const int b   = blockIdx.z;
    const int e0  = blockIdx.x * 64;
    const int h0  = blockIdx.y * 64;

    const int hp = (tid >> 3) * 2;       // h pair 0..62
    const int ec = (tid & 7) * 8;        // 8 contiguous e
    const float* s0 = enc + ((size_t)b * HDIM + h0 + hp) * TEDIM + e0 + ec;
    const float* s1 = s0 + TEDIM;
    float4 f0 = *(const float4*)(s0), f1 = *(const float4*)(s0 + 4);
    float4 g0 = *(const float4*)(s1), g1 = *(const float4*)(s1 + 4);
    *(unsigned int*)&tile[ec + 0][hp] = pack2(f0.x, g0.x);
    *(unsigned int*)&tile[ec + 1][hp] = pack2(f0.y, g0.y);
    *(unsigned int*)&tile[ec + 2][hp] = pack2(f0.z, g0.z);
    *(unsigned int*)&tile[ec + 3][hp] = pack2(f0.w, g0.w);
    *(unsigned int*)&tile[ec + 4][hp] = pack2(f1.x, g1.x);
    *(unsigned int*)&tile[ec + 5][hp] = pack2(f1.y, g1.y);
    *(unsigned int*)&tile[ec + 6][hp] = pack2(f1.z, g1.z);
    *(unsigned int*)&tile[ec + 7][hp] = pack2(f1.w, g1.w);
    __syncthreads();

    const int er = tid >> 2;             // 0..63
    const int hc = (tid & 3) * 16;       // 0..48
    unsigned short* dst = outT + ((size_t)b * TEDIM + e0 + er) * HDIM + h0 + hc;
    *(uint4*)(dst)     = *(const uint4*)&tile[er][hc];
    *(uint4*)(dst + 8) = *(const uint4*)&tile[er][hc + 8];
}

__device__ __forceinline__ void score_epilogue(
    const v8f& c, const float* __restrict__ oldsum, float* __restrict__ newsum_out,
    unsigned short attnS[][2056], float* racc, int b, int t0, int half, int ecol)
{
    #pragma unroll
    for (int r = 0; r < 8; ++r) {
        const int m = half * 8 + r;
        const size_t idx = ((size_t)b * TDIM + (t0 + m)) * TEDIM + ecol;
        const float ta  = __expf(c[r]);
        const float old = oldsum[idx];
        newsum_out[idx] = ta + old;
        const float at  = ta * __builtin_amdgcn_rcpf(old);
        attnS[m][ecol]  = bf1(at);
        racc[r] += at;
    }
}

// TIER: 0 = f32 only, 1 = bf16 copy (encbf), 2 = bf16 copy + bf16 transpose (encT)
template <int TIER>
__global__ __launch_bounds__(256)
void itta_fused_kernel(const float* __restrict__ hidden,            // (TD, B, H)
                       const float* __restrict__ enc,               // (B, H, TE) f32
                       const unsigned short* __restrict__ encbf,    // (B, H, TE) bf16
                       const unsigned short* __restrict__ encT,     // (B, TE, H) bf16
                       const float* __restrict__ oldsum,            // (B, TD, TE)
                       float* __restrict__ ctx_out,                 // (TD, B, H)
                       float* __restrict__ newsum_out)              // (B, TD, TE)
{
    __shared__ unsigned short hidA[16][520];     // bf16 hidden tile, 16 x H
    __shared__ unsigned short attnS[16][2056];   // bf16 un-normalized attention tile
    __shared__ float rowsum[16];

    const int tid  = threadIdx.x;
    const int lane = tid & 31;
    const int wave = tid >> 5;            // 0..7
    const int rA   = lane & 15;           // A row / B column / D column
    const int half = lane >> 4;           // 0 or 1
    const int t0   = blockIdx.x * 16;
    const int b    = blockIdx.y;

    // ---- Phase A: hidden tile (16 x 512 f32) -> bf16 LDS ----
    {
        const int m  = tid >> 4;
        const int hb = (tid & 15) * 32;
        const float* src = hidden + ((size_t)(t0 + m) * BATCH + b) * HDIM + hb;
        #pragma unroll
        for (int q = 0; q < 8; ++q) {
            float4 f = *(const float4*)(src + q * 4);
            *(unsigned int*)&hidA[m][hb + q*4 + 0] = pack2(f.x, f.y);
            *(unsigned int*)&hidA[m][hb + q*4 + 2] = pack2(f.z, f.w);
        }
    }
    if (tid < 16) rowsum[tid] = 0.0f;
    __syncthreads();

    float racc[8];
    #pragma unroll
    for (int r = 0; r < 8; ++r) racc[r] = 0.0f;

    // ---- Phase B: GEMM1 (scores) + exp + running-sum update ----
    if constexpr (TIER == 2) {
        // B-fragments straight from transposed bf16 global copy: no staging, no barriers.
        for (int et = 0; et < TEDIM / 128; ++et) {
            const int e0 = et * 128;
            v8f c = {};
            const unsigned short* bbase =
                encT + ((size_t)b * TEDIM + e0 + wave * 16 + rA) * HDIM + half * 16;
            #pragma unroll 4
            for (int ks = 0; ks < HDIM / 32; ++ks) {
                const int k0 = ks * 32;
                Frag a, bm;
                a.q[0]  = *(const uint4*)&hidA[rA][k0 + half * 16];
                a.q[1]  = *(const uint4*)&hidA[rA][k0 + half * 16 + 8];
                bm.q[0] = *(const uint4*)(bbase + k0);       // K = k0+half*16 .. +7
                bm.q[1] = *(const uint4*)(bbase + k0 + 8);   // K = .. +15
                c = __builtin_amdgcn_wmma_f32_16x16x32_bf16(
                        false, a.v, false, bm.v, (short)0, c, false, false);
            }
            score_epilogue(c, oldsum, newsum_out, attnS, racc,
                           b, t0, half, e0 + wave * 16 + rA);
        }
    } else {
        __shared__ unsigned short encB[128][136];   // e-major staging, 128e x 128h
        for (int et = 0; et < TEDIM / 128; ++et) {
            const int e0 = et * 128;
            v8f c = {};
            for (int kb = 0; kb < HDIM / 128; ++kb) {
                const int kbase = kb * 128;
                #pragma unroll
                for (int kk = 0; kk < 4; ++kk) {
                    const int hl = kk * 32 + (tid >> 4) * 2;    // local h pair
                    const int eb = (tid & 15) * 8;              // 8 contiguous e
                    if constexpr (TIER == 1) {
                        const unsigned short* s0 =
                            encbf + ((size_t)b * HDIM + (kbase + hl)) * TEDIM + e0 + eb;
                        const uint4 A  = *(const uint4*)(s0);
                        const uint4 Bv = *(const uint4*)(s0 + TEDIM);
                        *(unsigned int*)&encB[eb + 0][hl] = __builtin_amdgcn_perm(Bv.x, A.x, 0x05040100u);
                        *(unsigned int*)&encB[eb + 1][hl] = __builtin_amdgcn_perm(Bv.x, A.x, 0x07060302u);
                        *(unsigned int*)&encB[eb + 2][hl] = __builtin_amdgcn_perm(Bv.y, A.y, 0x05040100u);
                        *(unsigned int*)&encB[eb + 3][hl] = __builtin_amdgcn_perm(Bv.y, A.y, 0x07060302u);
                        *(unsigned int*)&encB[eb + 4][hl] = __builtin_amdgcn_perm(Bv.z, A.z, 0x05040100u);
                        *(unsigned int*)&encB[eb + 5][hl] = __builtin_amdgcn_perm(Bv.z, A.z, 0x07060302u);
                        *(unsigned int*)&encB[eb + 6][hl] = __builtin_amdgcn_perm(Bv.w, A.w, 0x05040100u);
                        *(unsigned int*)&encB[eb + 7][hl] = __builtin_amdgcn_perm(Bv.w, A.w, 0x07060302u);
                    } else {
                        const float* s0 = enc + ((size_t)b * HDIM + (kbase + hl)) * TEDIM + e0 + eb;
                        const float* s1 = s0 + TEDIM;
                        float4 a0 = *(const float4*)(s0);
                        float4 a1 = *(const float4*)(s0 + 4);
                        float4 b0 = *(const float4*)(s1);
                        float4 b1 = *(const float4*)(s1 + 4);
                        *(unsigned int*)&encB[eb + 0][hl] = pack2(a0.x, b0.x);
                        *(unsigned int*)&encB[eb + 1][hl] = pack2(a0.y, b0.y);
                        *(unsigned int*)&encB[eb + 2][hl] = pack2(a0.z, b0.z);
                        *(unsigned int*)&encB[eb + 3][hl] = pack2(a0.w, b0.w);
                        *(unsigned int*)&encB[eb + 4][hl] = pack2(a1.x, b1.x);
                        *(unsigned int*)&encB[eb + 5][hl] = pack2(a1.y, b1.y);
                        *(unsigned int*)&encB[eb + 6][hl] = pack2(a1.z, b1.z);
                        *(unsigned int*)&encB[eb + 7][hl] = pack2(a1.w, b1.w);
                    }
                }
                __syncthreads();
                #pragma unroll
                for (int ks = 0; ks < 4; ++ks) {
                    const int k0 = kbase + ks * 32;
                    const int kl = ks * 32;
                    Frag a, bm;
                    a.q[0]  = *(const uint4*)&hidA[rA][k0 + half * 16];
                    a.q[1]  = *(const uint4*)&hidA[rA][k0 + half * 16 + 8];
                    bm.q[0] = *(const uint4*)&encB[wave * 16 + rA][kl + half * 16];
                    bm.q[1] = *(const uint4*)&encB[wave * 16 + rA][kl + half * 16 + 8];
                    c = __builtin_amdgcn_wmma_f32_16x16x32_bf16(
                            false, a.v, false, bm.v, (short)0, c, false, false);
                }
                __syncthreads();
            }
            score_epilogue(c, oldsum, newsum_out, attnS, racc,
                           b, t0, half, e0 + wave * 16 + rA);
        }
    }

    #pragma unroll
    for (int r = 0; r < 8; ++r) atomicAdd(&rowsum[half * 8 + r], racc[r]);
    __syncthreads();

    float inv[8];
    #pragma unroll
    for (int r = 0; r < 8; ++r) inv[r] = __builtin_amdgcn_rcpf(rowsum[half * 8 + r]);

    // ---- Phase C: GEMM2  context = attn(16 x 2048) @ enc^T, normalize at store ----
    for (int i = 0; i < 4; ++i) {
        const int hcol = (wave * 4 + i) * 16 + rA;   // this lane's B column (h)
        v8f c = {};
        for (int ks = 0; ks < TEDIM / 32; ++ks) {
            const int k0 = ks * 32;
            Frag a, bm;
            a.q[0] = *(const uint4*)&attnS[rA][k0 + half * 16];
            a.q[1] = *(const uint4*)&attnS[rA][k0 + half * 16 + 8];
            if constexpr (TIER >= 1) {
                const unsigned short* bp =
                    encbf + ((size_t)b * HDIM + hcol) * TEDIM + half * 16 + k0;
                bm.q[0] = *(const uint4*)(bp);
                bm.q[1] = *(const uint4*)(bp + 8);
            } else {
                const float* bp = enc + ((size_t)b * HDIM + hcol) * TEDIM + half * 16 + k0;
                float4 f0 = *(const float4*)(bp);
                float4 f1 = *(const float4*)(bp + 4);
                float4 f2 = *(const float4*)(bp + 8);
                float4 f3 = *(const float4*)(bp + 12);
                bm.w[0] = pack2(f0.x, f0.y);
                bm.w[1] = pack2(f0.z, f0.w);
                bm.w[2] = pack2(f1.x, f1.y);
                bm.w[3] = pack2(f1.z, f1.w);
                bm.w[4] = pack2(f2.x, f2.y);
                bm.w[5] = pack2(f2.z, f2.w);
                bm.w[6] = pack2(f3.x, f3.y);
                bm.w[7] = pack2(f3.z, f3.w);
            }
            c = __builtin_amdgcn_wmma_f32_16x16x32_bf16(
                    false, a.v, false, bm.v, (short)0, c, false, false);
        }
        #pragma unroll
        for (int r = 0; r < 8; ++r) {
            const int m = half * 8 + r;
            ctx_out[((size_t)(t0 + m) * BATCH + b) * HDIM + hcol] = c[r] * inv[r];
        }
    }
}

extern "C" void kernel_launch(void* const* d_in, const int* in_sizes, int n_in,
                              void* d_out, int out_size, void* d_ws, size_t ws_size,
                              hipStream_t stream) {
    const float* hidden = (const float*)d_in[0];   // (TD, B, H)
    const float* enc    = (const float*)d_in[1];   // (B, H, TE)
    const float* oldsum = (const float*)d_in[2];   // (B, TD, TE)
    float* ctx    = (float*)d_out;                                  // (TD, B, H)
    float* newsum = (float*)d_out + (size_t)TDIM * BATCH * HDIM;    // (B, TD, TE)

    const size_t enc_elems = (size_t)BATCH * HDIM * TEDIM;
    const size_t bf16_sz   = enc_elems * sizeof(unsigned short);

    dim3 grid(TDIM / 16, BATCH);
    if (ws_size >= 2 * bf16_sz) {
        unsigned short* encbf = (unsigned short*)d_ws;
        unsigned short* encT  = (unsigned short*)((char*)d_ws + bf16_sz);
        conv_enc_kernel<<<(int)(enc_elems / 8 / 256), 256, 0, stream>>>(enc, encbf);
        conv_encT_kernel<<<dim3(TEDIM / 64, HDIM / 64, BATCH), 256, 0, stream>>>(enc, encT);
        itta_fused_kernel<2><<<grid, 256, 0, stream>>>(
            hidden, enc, encbf, encT, oldsum, ctx, newsum);
    } else if (ws_size >= bf16_sz) {
        unsigned short* encbf = (unsigned short*)d_ws;
        conv_enc_kernel<<<(int)(enc_elems / 8 / 256), 256, 0, stream>>>(enc, encbf);
        itta_fused_kernel<1><<<grid, 256, 0, stream>>>(
            hidden, enc, encbf, nullptr, oldsum, ctx, newsum);
    } else {
        itta_fused_kernel<0><<<grid, 256, 0, stream>>>(
            hidden, enc, nullptr, nullptr, oldsum, ctx, newsum);
    }
}
